// MentionEncoder_37950331027523
// MI455X (gfx1250) — matrix operand order repaired and checked by reading, not compiled, not run on hardware
//
#include <hip/hip_runtime.h>
#include <math.h>

// ---------------- CDNA5 WMMA types ----------------
typedef __attribute__((ext_vector_type(16))) __bf16 v16bf;
typedef __attribute__((ext_vector_type(8)))  __bf16 v8bf;
typedef __attribute__((ext_vector_type(8)))  float  v8f;

#define B_   8192
#define L_   8
#define T_   16
#define E_   512
#define D_   512
#define DC_  256

static __device__ inline v8f zero8() {
  v8f v;
#pragma unroll
  for (int i = 0; i < 8; ++i) v[i] = 0.0f;
  return v;
}

// A-matrix 16x32 bf16 fragment from a row-major f32 row.
// Lane layout (ISA 7.12.2): lanes 0-15: elems 0..7 = K0+kb..+7, elems 8..15 = K0+kb+16..+23
// with kb = 0 (lanes 0-15) or 8 (lanes 16-31).
static __device__ inline v16bf fraga_f32(const float* row, int K0, int kb) {
  const float4 a = *reinterpret_cast<const float4*>(row + K0 + kb);
  const float4 b = *reinterpret_cast<const float4*>(row + K0 + kb + 4);
  const float4 c = *reinterpret_cast<const float4*>(row + K0 + kb + 16);
  const float4 d = *reinterpret_cast<const float4*>(row + K0 + kb + 20);
  v16bf v;
  v[0]=(__bf16)a.x;  v[1]=(__bf16)a.y;  v[2]=(__bf16)a.z;  v[3]=(__bf16)a.w;
  v[4]=(__bf16)b.x;  v[5]=(__bf16)b.y;  v[6]=(__bf16)b.z;  v[7]=(__bf16)b.w;
  v[8]=(__bf16)c.x;  v[9]=(__bf16)c.y;  v[10]=(__bf16)c.z; v[11]=(__bf16)c.w;
  v[12]=(__bf16)d.x; v[13]=(__bf16)d.y; v[14]=(__bf16)d.z; v[15]=(__bf16)d.w;
  return v;
}

// A fragment from bf16 LDS row (same interleave).
static __device__ inline v16bf fraga_lds(const __bf16* row, int K0, int kb) {
  v16bf v;
#pragma unroll
  for (int j = 0; j < 8; ++j) {
    v[j]     = row[K0 + kb + j];
    v[8 + j] = row[K0 + kb + 16 + j];
  }
  return v;
}

// A fragment = bf16( lds_values + f32_pos ).
static __device__ inline v16bf fraga_attn(const __bf16* vrow, const float* prow,
                                          int K0, int kb) {
  v16bf v;
#pragma unroll
  for (int j = 0; j < 8; ++j) {
    v[j]     = (__bf16)((float)vrow[K0 + kb + j]      + prow[K0 + kb + j]);
    v[8 + j] = (__bf16)((float)vrow[K0 + kb + 16 + j] + prow[K0 + kb + 16 + j]);
  }
  return v;
}

// B-matrix 32x16 bf16 fragment from LDS: lane = column n; elems 0..7 = K0+kb..+7,
// elems 8..15 = K0+kb+16..+23 with kb = 0 (lanes 0-15) or 8 (lanes 16-31).
// Two 16B loads so only 16B alignment is required (lowered to ds_load_b128 x2).
static __device__ inline v16bf fragb_lds(const __bf16* p) {
  const v8bf lo = *reinterpret_cast<const v8bf*>(p);
  const v8bf hi = *reinterpret_cast<const v8bf*>(p + 16);
  v16bf v;
#pragma unroll
  for (int j = 0; j < 8; ++j) { v[j] = lo[j]; v[8 + j] = hi[j]; }
  return v;
}

// =====================================================================
// Kernel 1: word path. Fused embed-gather -> tanh GEMM -> (Wq-Wk) GEMM
// -> euclidean dist -> softmax -> weighted midpoint.
// Block = 128 threads (4 waves); each wave owns 16 rows = 2 mentions.
// Weight slab (16 out-rows x 512 K, bf16) shared by all 4 waves.
// =====================================================================
__global__ void __launch_bounds__(128) mention_word_kernel(
    const int* __restrict__ mentions,
    const float* __restrict__ word_lut,
    const float* __restrict__ W_word, const float* __restrict__ b_word,
    const float* __restrict__ pos_embeds,
    const float* __restrict__ Wq, const float* __restrict__ bq,
    const float* __restrict__ Wk, const float* __restrict__ bk,
    const float* __restrict__ beta_p,
    float* __restrict__ out_mention) {
  constexpr int VSTR = D_ + 16;
  __shared__ __attribute__((aligned(32))) __bf16 s_wslab[16 * D_];   // 16KB slab
  __shared__ __attribute__((aligned(32))) __bf16 s_vals[4][16][VSTR];
  __shared__ float s_dist[4][16];
  __shared__ float s_wt[4][16];

  const int tid  = threadIdx.x;
  const int wave = tid >> 5;
  const int lane = tid & 31;
  const int m    = lane & 15;   // row (A) / column (B,C,D) index within tile
  const int half = lane >> 4;
  const int kb   = half * 8;

  const int rowbase = blockIdx.x * 64 + wave * 16;
  const int row_m   = rowbase + m;
  const int tok     = mentions[row_m];
  const float* erow = word_lut + (size_t)tok * E_;

  // A fragments: this lane's embedding row, all 16 K-steps (held in VGPRs).
  v16bf af[16];
#pragma unroll
  for (int ks = 0; ks < 16; ++ks) af[ks] = fraga_f32(erow, ks * 32, kb);

  // ---------------- GEMM1: values = tanh(E @ W_word^T + b_word) ----------------
  for (int nt = 0; nt < 32; ++nt) {
    __syncthreads();
    for (int i = tid * 4; i < 16 * D_; i += 128 * 4) {
      const int r = i >> 9;
      const int c = i & (D_ - 1);
      const float4 w =
          *reinterpret_cast<const float4*>(W_word + (size_t)(nt * 16 + r) * E_ + c);
      s_wslab[i + 0] = (__bf16)w.x;
      s_wslab[i + 1] = (__bf16)w.y;
      s_wslab[i + 2] = (__bf16)w.z;
      s_wslab[i + 3] = (__bf16)w.w;
    }
    if (nt < 31 && tid < 64)  // gfx1250 global_prefetch_b8 of the next slab
      __builtin_prefetch(W_word + (size_t)((nt + 1) * 16) * E_ + tid * 128, 0, 0);
    __syncthreads();

    // Prefetch all 16 B fragments into VGPRs, then run the WMMA chain so
    // ds_load latency overlaps the matrix pipe (partial s_wait_dscnt).
    const __bf16* brow = &s_wslab[m * D_];
    v16bf bfr[16];
#pragma unroll
    for (int ks = 0; ks < 16; ++ks) bfr[ks] = fragb_lds(brow + ks * 32 + half * 8);
    v8f acc = zero8();
#pragma unroll
    for (int ks = 0; ks < 16; ++ks)
      acc = __builtin_amdgcn_wmma_f32_16x16x32_bf16(
          false, af[ks], false, bfr[ks], (short)0, acc, false, false);

    const int n_g = nt * 16 + m;
    const float bias = b_word[n_g];
#pragma unroll
    for (int r = 0; r < 8; ++r) {
      const float v = tanhf(acc[r] + bias);
      s_vals[wave][r + half * 8][n_g] = (__bf16)v;  // lane = column, reg = row
    }
  }

  // A fragments for attention: bf16(values + pos_embed)   (intra-wave LDS, in-order)
  const int pidx = (tok > 0) ? ((row_m & (L_ - 1)) + 1) : 0;
  const float* prow  = pos_embeds + (size_t)pidx * D_;
  const __bf16* vrow = &s_vals[wave][m][0];
#pragma unroll
  for (int ks = 0; ks < 16; ++ks) af[ks] = fraga_attn(vrow, prow, ks * 32, kb);

  // ---------------- GEMM2: diff = A @ (Wq-Wk)^T + (bq-bk); dist^2 ----------------
  float ssq[8] = {0.f, 0.f, 0.f, 0.f, 0.f, 0.f, 0.f, 0.f};
  for (int nt = 0; nt < 32; ++nt) {
    __syncthreads();
    for (int i = tid * 4; i < 16 * D_; i += 128 * 4) {
      const int r = i >> 9;
      const int c = i & (D_ - 1);
      const size_t widx = (size_t)(nt * 16 + r) * D_ + c;
      const float4 q = *reinterpret_cast<const float4*>(Wq + widx);
      const float4 k = *reinterpret_cast<const float4*>(Wk + widx);
      s_wslab[i + 0] = (__bf16)(q.x - k.x);
      s_wslab[i + 1] = (__bf16)(q.y - k.y);
      s_wslab[i + 2] = (__bf16)(q.z - k.z);
      s_wslab[i + 3] = (__bf16)(q.w - k.w);
    }
    __syncthreads();

    const __bf16* brow = &s_wslab[m * D_];
    v16bf bfr[16];
#pragma unroll
    for (int ks = 0; ks < 16; ++ks) bfr[ks] = fragb_lds(brow + ks * 32 + half * 8);
    v8f acc = zero8();
#pragma unroll
    for (int ks = 0; ks < 16; ++ks)
      acc = __builtin_amdgcn_wmma_f32_16x16x32_bf16(
          false, af[ks], false, bfr[ks], (short)0, acc, false, false);

    const int n_g = nt * 16 + m;
    const float bd = bq[n_g] - bk[n_g];
#pragma unroll
    for (int r = 0; r < 8; ++r) {
      const float dv = acc[r] + bd;
      ssq[r] += dv * dv;
    }
  }

  // Cross-lane reduce within each 16-lane half: lanes 0-15 carry rows 0-7,
  // lanes 16-31 carry rows 8-15.
#pragma unroll
  for (int r = 0; r < 8; ++r) {
    float s = ssq[r];
    s += __shfl_xor(s, 1);
    s += __shfl_xor(s, 2);
    s += __shfl_xor(s, 4);
    s += __shfl_xor(s, 8);
    ssq[r] = s;
  }
  if (lane == 0 || lane == 16) {
#pragma unroll
    for (int r = 0; r < 8; ++r) s_dist[wave][half * 8 + r] = sqrtf(ssq[r]);
  }

  // Softmax over the 8 positions of each mention (2 mentions per wave).
  const float beta = beta_p[0];
  if (half == 0 && (m == 0 || m == 8)) {
    const int base = m;  // 0 or 8
    float mx = -1e30f;
#pragma unroll
    for (int i = 0; i < 8; ++i) {
      const float z = -beta * s_dist[wave][base + i];
      mx = fmaxf(mx, z);
    }
    float e[8], ssum = 0.f;
#pragma unroll
    for (int i = 0; i < 8; ++i) {
      e[i] = expf(-beta * s_dist[wave][base + i] - mx);
      ssum += e[i];
    }
    const float inv = 1.0f / ssum;
#pragma unroll
    for (int i = 0; i < 8; ++i) s_wt[wave][base + i] = e[i] * inv;
  }

  // Weighted euclidean midpoint (softmax weights sum to 1).
#pragma unroll
  for (int men = 0; men < 2; ++men) {
    const int gm = (rowbase >> 3) + men;
    for (int d = lane; d < D_; d += 32) {
      float a = 0.f;
#pragma unroll
      for (int i = 0; i < 8; ++i)
        a += s_wt[wave][men * 8 + i] * (float)s_vals[wave][men * 8 + i][d];
      out_mention[(size_t)gm * D_ + d] = a;
    }
  }
}

// =====================================================================
// Kernel 2: char RNN, fully fused over T=16 steps.
// Block = 128 threads (4 waves); each wave owns 16 batch rows.
// CDNA5 big-LDS: both 256x256 weight matrices live in LDS as bf16
// (264KB + 33KB h-buffer < 320KB/WGP), converted once per block.
// t=0 peeled; steady state runs a branch-free 16-WMMA chain per tile.
// =====================================================================
__global__ void __launch_bounds__(128) char_rnn_kernel(
    const int* __restrict__ chars,
    const float* __restrict__ char_lut,
    const float* __restrict__ W_rnn, const float* __restrict__ U_rnn,
    const float* __restrict__ b_rnn,
    float* __restrict__ out_char) {
  constexpr int WSTR = DC_ + 8;  // 264: 16B-aligned rows, conflict-free b128 reads
  __shared__ __attribute__((aligned(32))) __bf16 s_wu[2][DC_][WSTR];  // [0]=U, [1]=W
  __shared__ __attribute__((aligned(32))) __bf16 s_h[4][16][WSTR];

  const int tid  = threadIdx.x;
  const int wave = tid >> 5;
  const int lane = tid & 31;
  const int m    = lane & 15;
  const int half = lane >> 4;
  const int kb   = half * 8;

  // Convert both weight matrices f32 -> bf16 into LDS, once per block.
  for (int i = tid * 4; i < DC_ * DC_; i += 128 * 4) {
    const int r = i >> 8;
    const int c = i & (DC_ - 1);
    const float4 u = *reinterpret_cast<const float4*>(U_rnn + i);
    const float4 w = *reinterpret_cast<const float4*>(W_rnn + i);
    s_wu[0][r][c + 0] = (__bf16)u.x;
    s_wu[0][r][c + 1] = (__bf16)u.y;
    s_wu[0][r][c + 2] = (__bf16)u.z;
    s_wu[0][r][c + 3] = (__bf16)u.w;
    s_wu[1][r][c + 0] = (__bf16)w.x;
    s_wu[1][r][c + 1] = (__bf16)w.y;
    s_wu[1][r][c + 2] = (__bf16)w.z;
    s_wu[1][r][c + 3] = (__bf16)w.w;
  }
  __syncthreads();

  const int rowbase = blockIdx.x * 64 + wave * 16;
  const int row_m   = rowbase + m;
  const int* cids   = chars + (size_t)row_m * T_;

  v8f hsum[16];
#pragma unroll
  for (int nt = 0; nt < 16; ++nt) hsum[nt] = zero8();

  v16bf afH[8], afX[8];

  // ---------------- t = 0: h0 == 0, so h1 = tanh(x0 @ U^T + b) ----------------
  {
    const int c0 = cids[0];
    const float* xrow = char_lut + (size_t)c0 * DC_;
#pragma unroll
    for (int ks = 0; ks < 8; ++ks) afX[ks] = fraga_f32(xrow, ks * 32, kb);

#pragma unroll
    for (int nt = 0; nt < 16; ++nt) {
      const int n_g = nt * 16 + m;
      v16bf bfr[8];
#pragma unroll
      for (int ks = 0; ks < 8; ++ks) bfr[ks] = fragb_lds(&s_wu[0][n_g][ks * 32 + half * 8]);
      v8f acc = zero8();
#pragma unroll
      for (int ks = 0; ks < 8; ++ks)
        acc = __builtin_amdgcn_wmma_f32_16x16x32_bf16(
            false, afX[ks], false, bfr[ks], (short)0, acc, false, false);
      const float bias = b_rnn[n_g];
#pragma unroll
      for (int r = 0; r < 8; ++r) {
        const float h = tanhf(acc[r] + bias);
        hsum[nt][r] += h;
        s_h[wave][r + half * 8][n_g] = (__bf16)h;
      }
    }
  }

  // ---------------- t = 1..15: h = tanh(x@U^T + h@W^T + b) ----------------
  for (int t = 1; t < T_; ++t) {
    const int c = cids[t];
    const float* xrow = char_lut + (size_t)c * DC_;
#pragma unroll
    for (int ks = 0; ks < 8; ++ks) afX[ks] = fraga_f32(xrow, ks * 32, kb);

    const __bf16* hrow = &s_h[wave][m][0];  // intra-wave LDS, in-order DS
#pragma unroll
    for (int ks = 0; ks < 8; ++ks) afH[ks] = fraga_lds(hrow, ks * 32, kb);

#pragma unroll
    for (int nt = 0; nt < 16; ++nt) {
      const int n_g = nt * 16 + m;
      v16bf bu[8], bw[8];
#pragma unroll
      for (int ks = 0; ks < 8; ++ks) {
        bu[ks] = fragb_lds(&s_wu[0][n_g][ks * 32 + half * 8]);
        bw[ks] = fragb_lds(&s_wu[1][n_g][ks * 32 + half * 8]);
      }
      v8f acc = zero8();
#pragma unroll
      for (int ks = 0; ks < 8; ++ks)
        acc = __builtin_amdgcn_wmma_f32_16x16x32_bf16(
            false, afX[ks], false, bu[ks], (short)0, acc, false, false);
#pragma unroll
      for (int ks = 0; ks < 8; ++ks)
        acc = __builtin_amdgcn_wmma_f32_16x16x32_bf16(
            false, afH[ks], false, bw[ks], (short)0, acc, false, false);

      const float bias = b_rnn[n_g];
#pragma unroll
      for (int r = 0; r < 8; ++r) {
        const float h = tanhf(acc[r] + bias);
        hsum[nt][r] += h;
        s_h[wave][r + half * 8][n_g] = (__bf16)h;  // for next step's A frags
      }
    }
  }

  // char_out = mean over T of h_t
#pragma unroll
  for (int nt = 0; nt < 16; ++nt) {
    const int n_g = nt * 16 + m;
#pragma unroll
    for (int r = 0; r < 8; ++r) {
      const int row = rowbase + r + half * 8;
      out_char[(size_t)row * DC_ + n_g] = hsum[nt][r] * (1.0f / (float)T_);
    }
  }
}

// =====================================================================
extern "C" void kernel_launch(void* const* d_in, const int* in_sizes, int n_in,
                              void* d_out, int out_size, void* d_ws, size_t ws_size,
                              hipStream_t stream) {
  (void)in_sizes; (void)n_in; (void)out_size; (void)d_ws; (void)ws_size;
  const int*   mentions = (const int*)d_in[0];
  const int*   mchars   = (const int*)d_in[1];
  const float* word_lut = (const float*)d_in[2];
  const float* W_word   = (const float*)d_in[3];
  const float* b_word   = (const float*)d_in[4];
  const float* pos_emb  = (const float*)d_in[5];
  const float* Wq       = (const float*)d_in[6];
  const float* bq       = (const float*)d_in[7];
  const float* Wk       = (const float*)d_in[8];
  const float* bk       = (const float*)d_in[9];
  const float* beta     = (const float*)d_in[10];
  const float* char_lut = (const float*)d_in[11];
  const float* W_rnn    = (const float*)d_in[12];
  const float* U_rnn    = (const float*)d_in[13];
  const float* b_rnn    = (const float*)d_in[14];
  float* out = (float*)d_out;

  // Word path: 65536 rows / 64 rows per block
  mention_word_kernel<<<(B_ * L_) / 64, 128, 0, stream>>>(
      mentions, word_lut, W_word, b_word, pos_emb, Wq, bq, Wk, bk, beta, out);

  // Char RNN: 8192 rows / 64 rows per block
  char_rnn_kernel<<<B_ / 64, 128, 0, stream>>>(
      mchars, char_lut, W_rnn, U_rnn, b_rnn, out + (size_t)B_ * D_);
}
